// GCNEdgeClassifier_47897475285650
// MI455X (gfx1250) — matrix operand order repaired and checked by reading, not compiled, not run on hardware
//
#include <hip/hip_runtime.h>
#include <hip/hip_bf16.h>

#define N_NODES 50000
#define N_EDGES 800000
#define HID 64
#define ER_PAD 160   // 153 padded to multiple of 32

typedef _Float16 v16h __attribute__((ext_vector_type(16)));
typedef _Float16 v8h  __attribute__((ext_vector_type(8)));
typedef float    v8f  __attribute__((ext_vector_type(8)));

// ---------------- degree / norm ----------------
__global__ void deg_kernel(const int* __restrict__ dst, float* __restrict__ deg) {
    int e = blockIdx.x * blockDim.x + threadIdx.x;
    if (e < N_EDGES) atomicAdd(&deg[dst[e]], 1.0f);
}

__global__ void norm_kernel(float* __restrict__ norm) {
    int i = blockIdx.x * blockDim.x + threadIdx.x;
    if (i < N_NODES) {
        float d = norm[i];
        d = d < 1.0f ? 1.0f : d;
        norm[i] = rsqrtf(d);
    }
}

// ---------------- weight conversion: f32 (K x 64) -> f16 N-major (64 x ldk) ----------------
__global__ void convw_kernel(const float* __restrict__ M0, const float* __restrict__ M1,
                             _Float16* __restrict__ M0t, _Float16* __restrict__ M1t) {
    int t = blockIdx.x * blockDim.x + threadIdx.x;
    if (t < 64 * ER_PAD) {
        int n = t / ER_PAD, k = t % ER_PAD;
        M0t[t] = (k < 153) ? (_Float16)M0[k * 64 + n] : (_Float16)0.0f;
    }
    int t2 = t - 64 * ER_PAD;
    if (t2 >= 0 && t2 < 64 * 64) {
        int n = t2 / 64, k = t2 % 64;
        M1t[t2] = (_Float16)M1[k * 64 + n];
    }
}

// ---------------- GCN: per-node Wh*norm, scatter, finalize ----------------
__global__ void nodew_kernel(const float* __restrict__ hin, int K,
                             const float* __restrict__ W, const float* __restrict__ b,
                             const float* __restrict__ norm, float* __restrict__ whn) {
    int t = blockIdx.x * blockDim.x + threadIdx.x;
    if (t >= N_NODES * 64) return;
    int n = t >> 6, j = t & 63;
    const float* hr = hin + (long long)n * K;
    float acc = b[j];
    for (int k = 0; k < K; ++k) acc += hr[k] * W[k * 64 + j];
    whn[t] = acc * norm[n];
}

__global__ void scatter_kernel(const int* __restrict__ src, const int* __restrict__ dst,
                               const float* __restrict__ whn, float* __restrict__ agg) {
    long long t = (long long)blockIdx.x * blockDim.x + threadIdx.x;
    int e = (int)(t >> 6);
    int j = (int)(t & 63);
    if (e < N_EDGES)
        atomicAdd(&agg[(long long)dst[e] * 64 + j], whn[(long long)src[e] * 64 + j]);
}

__global__ void finalize_kernel(const float* __restrict__ agg, const float* __restrict__ norm,
                                float* __restrict__ h, int residual) {
    int t = blockIdx.x * blockDim.x + threadIdx.x;
    if (t >= N_NODES * 64) return;
    int n = t >> 6;
    float v = agg[t] * norm[n];
    v = v > 0.0f ? v : 0.0f;
    h[t] = residual ? (v + h[t]) : v;
}

// ---------------- WMMA fragment loaders (CDNA5 16x16x32 f16 layouts) ----------------
// A (16xK tile, row-major f16, stride ldk): lane<16 -> K {k0..k0+15},
// lane>=16 -> +16 offset; row = lane&15.
__device__ inline v16h load_a_frag(const _Float16* base, int ldk, int lane, int k0) {
    const _Float16* rp = base + (lane & 15) * ldk + k0 + ((lane & 16) ? 16 : 0);
    v8h lo = *(const v8h*)rp;
    v8h hi = *(const v8h*)(rp + 8);
    v16h a;
#pragma unroll
    for (int i = 0; i < 8; ++i) { a[i] = lo[i]; a[8 + i] = hi[i]; }
    return a;
}
// B (Kx16 tile from N-major weights Wt[64][ldk]): lane holds column n0+(lane&15),
// 16 contiguous K starting at k0 (+16 for upper half-wave).
__device__ inline v16h load_b_frag(const _Float16* wt, int ldk, int lane, int n0, int k0) {
    const _Float16* rp = wt + (n0 + (lane & 15)) * ldk + k0 + ((lane & 16) ? 16 : 0);
    return *(const v16h*)rp;
}

// ---------------- edge MLP: 16 edges per wave, 4 waves per block ----------------
__global__ __launch_bounds__(128) void edge_mlp_kernel(
    const float* __restrict__ h,
    const int* __restrict__ src, const int* __restrict__ dst,
    const int* __restrict__ pc, const int* __restrict__ rc, const int* __restrict__ pf,
    const float* __restrict__ nums,
    const float* __restrict__ curr_emb, const float* __restrict__ pay_emb,
    const _Float16* __restrict__ M0t, const float* __restrict__ bM0,
    const _Float16* __restrict__ M1t, const float* __restrict__ bM1,
    const float* __restrict__ M2, const float* __restrict__ bM2,
    float* __restrict__ out)
{
    __shared__ __align__(32) _Float16 A[4][16][ER_PAD];
    __shared__ __align__(32) _Float16 X1[4][16][HID];
    __shared__ __align__(32) _Float16 X2[4][16][HID];

    const int lane = threadIdx.x & 31;
    const int w    = threadIdx.x >> 5;
    const long long e0 = ((long long)blockIdx.x * 4 + w) * 16;

    // ---- build 16x160 edge-repr tile in LDS (each lane fills half a row) ----
    {
        int r = lane & 15;
        long long e = e0 + r;
        const float* hs  = h + (long long)src[e] * 64;
        const float* hd  = h + (long long)dst[e] * 64;
        const float* cep = curr_emb + pc[e] * 8;
        const float* cer = curr_emb + rc[e] * 8;
        const float* pe  = pay_emb + pf[e] * 4;
        const float* nm  = nums + e * 5;
        int c0 = (lane >> 4) * 80;
        for (int c = c0; c < c0 + 80; ++c) {
            float v;
            if      (c < 64)  v = hs[c];
            else if (c < 128) v = hd[c - 64];
            else if (c < 136) v = cep[c - 128];
            else if (c < 144) v = cer[c - 136];
            else if (c < 148) v = pe[c - 144];
            else if (c < 153) v = nm[c - 148];
            else              v = 0.0f;
            A[w][r][c] = (_Float16)v;
        }
    }
    const int mhi = (lane & 16) ? 8 : 0;
    const int nc  = lane & 15;

    // ---- layer 0: (16x160) @ (160x64) + bM0, relu -> X1 ----
#pragma unroll
    for (int nt = 0; nt < 4; ++nt) {
        v8f c = {};
#pragma unroll
        for (int k0 = 0; k0 < ER_PAD; k0 += 32) {
            v16h a = load_a_frag(&A[w][0][0], ER_PAD, lane, k0);
            v16h b = load_b_frag(M0t, ER_PAD, lane, nt * 16, k0);
            c = __builtin_amdgcn_wmma_f32_16x16x32_f16(false, a, false, b, (short)0, c, false, false);
        }
        float bias = bM0[nt * 16 + nc];
#pragma unroll
        for (int r = 0; r < 8; ++r) {
            float v = c[r] + bias;
            X1[w][r + mhi][nt * 16 + nc] = (_Float16)(v > 0.0f ? v : 0.0f);
        }
    }

    // ---- layer 1: (16x64) @ (64x64) + bM1, relu -> X2 ----
#pragma unroll
    for (int nt = 0; nt < 4; ++nt) {
        v8f c = {};
#pragma unroll
        for (int k0 = 0; k0 < HID; k0 += 32) {
            v16h a = load_a_frag(&X1[w][0][0], HID, lane, k0);
            v16h b = load_b_frag(M1t, HID, lane, nt * 16, k0);
            c = __builtin_amdgcn_wmma_f32_16x16x32_f16(false, a, false, b, (short)0, c, false, false);
        }
        float bias = bM1[nt * 16 + nc];
#pragma unroll
        for (int r = 0; r < 8; ++r) {
            float v = c[r] + bias;
            X2[w][r + mhi][nt * 16 + nc] = (_Float16)(v > 0.0f ? v : 0.0f);
        }
    }

    // ---- head: (16x64) @ (64x2) + bM2 -> out; 32 lanes = 16 rows x 2 outputs ----
    {
        int r = lane & 15;
        int o = lane >> 4;
        float acc = bM2[o];
        for (int k = 0; k < 64; ++k)
            acc += (float)X2[w][r][k] * M2[k * 2 + o];
        out[(e0 + r) * 2 + o] = acc;
    }
}

// ---------------- host launcher ----------------
extern "C" void kernel_launch(void* const* d_in, const int* in_sizes, int n_in,
                              void* d_out, int out_size, void* d_ws, size_t ws_size,
                              hipStream_t stream) {
    const float* node_feats = (const float*)d_in[0];
    const int*   src        = (const int*)d_in[1];
    const int*   dst        = (const int*)d_in[2];
    const int*   pc         = (const int*)d_in[3];
    const int*   rc         = (const int*)d_in[4];
    const int*   pf         = (const int*)d_in[5];
    const float* nums       = (const float*)d_in[6];
    const float* W0         = (const float*)d_in[7];
    const float* b0         = (const float*)d_in[8];
    const float* W1         = (const float*)d_in[9];
    const float* b1         = (const float*)d_in[10];
    const float* W2         = (const float*)d_in[11];
    const float* b2         = (const float*)d_in[12];
    const float* curr_emb   = (const float*)d_in[13];
    const float* pay_emb    = (const float*)d_in[14];
    const float* M0         = (const float*)d_in[15];
    const float* bM0        = (const float*)d_in[16];
    const float* M1         = (const float*)d_in[17];
    const float* bM1        = (const float*)d_in[18];
    const float* M2         = (const float*)d_in[19];
    const float* bM2        = (const float*)d_in[20];

    float* ws   = (float*)d_ws;
    float* norm = ws;                                   // N_NODES (deg then norm)
    float* h    = norm + N_NODES;                       // N_NODES*64
    float* whn  = h + (size_t)N_NODES * 64;             // N_NODES*64
    float* agg  = whn + (size_t)N_NODES * 64;           // N_NODES*64
    _Float16* M0t = (_Float16*)(agg + (size_t)N_NODES * 64);  // 64*160 f16 (32B aligned)
    _Float16* M1t = M0t + 64 * ER_PAD;                        // 64*64 f16

    const int T = 256;
    // degree + norm
    hipMemsetAsync(norm, 0, N_NODES * sizeof(float), stream);
    deg_kernel<<<(N_EDGES + T - 1) / T, T, 0, stream>>>(dst, norm);
    norm_kernel<<<(N_NODES + T - 1) / T, T, 0, stream>>>(norm);
    // weight conversion
    convw_kernel<<<(64 * ER_PAD + 64 * 64 + T - 1) / T, T, 0, stream>>>(M0, M1, M0t, M1t);

    const int nodeBlocks = (N_NODES * 64 + T - 1) / T;
    const int scatBlocks = (int)(((long long)N_EDGES * 64 + T - 1) / T);

    // GCN layer 0 (K=2, no residual)
    nodew_kernel<<<nodeBlocks, T, 0, stream>>>(node_feats, 2, W0, b0, norm, whn);
    hipMemsetAsync(agg, 0, (size_t)N_NODES * 64 * sizeof(float), stream);
    scatter_kernel<<<scatBlocks, T, 0, stream>>>(src, dst, whn, agg);
    finalize_kernel<<<nodeBlocks, T, 0, stream>>>(agg, norm, h, 0);

    // GCN layer 1 (K=64, residual)
    nodew_kernel<<<nodeBlocks, T, 0, stream>>>(h, 64, W1, b1, norm, whn);
    hipMemsetAsync(agg, 0, (size_t)N_NODES * 64 * sizeof(float), stream);
    scatter_kernel<<<scatBlocks, T, 0, stream>>>(src, dst, whn, agg);
    finalize_kernel<<<nodeBlocks, T, 0, stream>>>(agg, norm, h, 1);

    // GCN layer 2 (K=64, residual)
    nodew_kernel<<<nodeBlocks, T, 0, stream>>>(h, 64, W2, b2, norm, whn);
    hipMemsetAsync(agg, 0, (size_t)N_NODES * 64 * sizeof(float), stream);
    scatter_kernel<<<scatBlocks, T, 0, stream>>>(src, dst, whn, agg);
    finalize_kernel<<<nodeBlocks, T, 0, stream>>>(agg, norm, h, 1);

    // edge MLP: 16 edges/wave, 4 waves/block -> 800000/64 = 12500 blocks
    edge_mlp_kernel<<<N_EDGES / 64, 128, 0, stream>>>(
        h, src, dst, pc, rc, pf, nums, curr_emb, pay_emb,
        M0t, bM0, M1t, bM1, M2, bM2, (float*)d_out);
}
